// TimeSformer_49692771615119
// MI455X (gfx1250) — matrix operand order repaired and checked by reading, not compiled, not run on hardware
//
#include <hip/hip_runtime.h>

// ---------------- problem constants (fixed by setup_inputs) ----------------
constexpr int BATCH = 8;
constexpr int FRM   = 16;      // f
constexpr int NTOK  = 196;     // n (spatial tokens per frame)
constexpr int HEADS = 16;      // H
constexpr int DH    = 64;      // head dim
constexpr int DIM   = 1024;    // model dim
constexpr int NSEQ  = 1 + FRM * NTOK;      // 3137 tokens
constexpr int BH    = BATCH * HEADS;       // 128
constexpr int MROWS = BATCH * NSEQ;        // 25096
constexpr int QKVN  = 3 * HEADS * DH;      // 3072
constexpr int MT    = (MROWS + 15) / 16;   // 1569 m-tiles
constexpr int KP    = 224;                 // padded key count (197 -> 14 tiles)
constexpr int QP    = 208;                 // padded query rows (196 -> 13 tiles)
constexpr int KCH   = 256;                 // GEMM K-chunk staged in LDS

// ---------------- WMMA types ----------------
typedef __bf16    v16bf __attribute__((ext_vector_type(16)));
typedef float     v8f   __attribute__((ext_vector_type(8)));
typedef unsigned  u32x4 __attribute__((ext_vector_type(4)));
typedef int       vint4 __attribute__((vector_size(4 * sizeof(int))));

union BFrag16 { v16bf v; u32x4 q[2]; };

__device__ __forceinline__ unsigned short f2bf(float f) {
    union { float f; unsigned u; } v; v.f = f;
    unsigned u = v.u;
    unsigned r = (u + 0x7FFFu + ((u >> 16) & 1u)) >> 16;   // RNE
    return (unsigned short)r;
}
__device__ __forceinline__ float bf2f(unsigned short h) {
    union { unsigned u; float f; } v; v.u = ((unsigned)h) << 16;
    return v.f;
}

__device__ __forceinline__ v8f wmma_bf16(const BFrag16& a, const BFrag16& b, v8f c) {
    return __builtin_amdgcn_wmma_f32_16x16x32_bf16(false, a.v, false, b.v,
                                                   (short)0, c, false, false);
}

// A fragment (16x32 bf16, M x K): lane m=l&15, halves hold K {0..7|8..15} and {16..23|24..31}
__device__ __forceinline__ void load_a(BFrag16& fr, const unsigned short* row,
                                       int kk, int hi) {
    fr.q[0] = *reinterpret_cast<const u32x4*>(row + kk + hi * 8);
    fr.q[1] = *reinterpret_cast<const u32x4*>(row + kk + 16 + hi * 8);
}
// B fragment (32x16 bf16, K x N) from N-major memory (row = n, K contiguous):
// lane n=l&15, halves hold K {0..7|8..15} and {16..23|24..31}
__device__ __forceinline__ void load_b(BFrag16& fr, const unsigned short* nrow,
                                       int kk, int hi) {
    const unsigned short* p = nrow + kk + hi * 8;
    fr.q[0] = *reinterpret_cast<const u32x4*>(p);
    fr.q[1] = *reinterpret_cast<const u32x4*>(p + 16);
}

// ------------- async global->LDS copy (CDNA5), with safe fallback ----------
#if __has_builtin(__builtin_amdgcn_global_load_async_to_lds_b128)
#define HAS_ASYNC_LDS 1
#else
#define HAS_ASYNC_LDS 0
#endif

__device__ __forceinline__ void async_cp16(const unsigned short* g, unsigned short* l) {
#if HAS_ASYNC_LDS
    __builtin_amdgcn_global_load_async_to_lds_b128((vint4*)g, (vint4*)l, 0, 0);
#else
    *reinterpret_cast<u32x4*>(l) = *reinterpret_cast<const u32x4*>(g);
#endif
}
__device__ __forceinline__ void async_wait0() {
#if HAS_ASYNC_LDS
#if __has_builtin(__builtin_amdgcn_s_wait_asynccnt)
    __builtin_amdgcn_s_wait_asynccnt(0);
#else
    asm volatile("s_wait_asynccnt 0" ::: "memory");
#endif
#endif
}

// ---------------- fp32 -> bf16 convert ----------------
__global__ __launch_bounds__(256) void cvt_bf16(const float* __restrict__ in,
                                                unsigned short* __restrict__ out, int n) {
    for (int i = blockIdx.x * blockDim.x + threadIdx.x; i < n; i += gridDim.x * blockDim.x)
        out[i] = f2bf(in[i]);
}

// ==================== shared GEMM body =====================================
// Block = 8 waves; wave w owns m-tile (blockIdx.x*8+w); block owns 64 N-cols.
// Weight slice (64 x DIM) staged in LDS in KCH-chunks, double buffered, async.
// Inner loop: A and all 4 B fragments software-pipelined one k-step ahead.
template <typename EPI>
__device__ __forceinline__ void gemm_body(const unsigned short* __restrict__ ab,
                                          const unsigned short* __restrict__ wb,
                                          char* smem, EPI epilogue) {
    unsigned short* bbuf[2] = { (unsigned short*)smem,
                                (unsigned short*)smem + 64 * KCH };
    const int tid  = threadIdx.x;
    const int lane = tid & 31, wave = tid >> 5;
    const int ln = lane & 15, hi = lane >> 4;
    int mt = blockIdx.x * 8 + wave;
    const bool valid = (mt < MT);
    if (!valid) mt = MT - 1;
    const int mbase = mt * 16;
    const int nbase = blockIdx.y * 64;

    int mrow = mbase + ln; if (mrow >= MROWS) mrow = MROWS - 1;
    const unsigned short* arow = ab + (size_t)mrow * DIM;

    // cooperative async stage of one 64 x KCH weight chunk (32KB)
    auto stage = [&](int c, int buf) {
        const unsigned short* wsrc = wb + (size_t)nbase * DIM + c * KCH;
        unsigned short* dst = bbuf[buf];
#pragma unroll
        for (int u = 0; u < 8; ++u) {
            const int idx = tid + u * 256;
            const int row = idx >> 5, cc = idx & 31;
            async_cp16(wsrc + (size_t)row * DIM + cc * 8, dst + row * KCH + cc * 8);
        }
    };

    v8f acc[4];
#pragma unroll
    for (int t = 0; t < 4; ++t) acc[t] = v8f{0,0,0,0,0,0,0,0};

    stage(0, 0);
    async_wait0();
    __syncthreads();

    constexpr int NCHUNK = DIM / KCH;
    for (int c = 0; c < NCHUNK; ++c) {
        if (c + 1 < NCHUNK) stage(c + 1, (c + 1) & 1);
        const unsigned short* bufc = bbuf[c & 1];
        const unsigned short* arowc = arow + c * KCH;
        // double-buffered fragments: load step kk+32 while multiplying step kk
        BFrag16 abuf[2];
        BFrag16 bb[2][4];
        load_a(abuf[0], arowc, 0, hi);
#pragma unroll
        for (int t = 0; t < 4; ++t) load_b(bb[0][t], bufc + (t * 16 + ln) * KCH, 0, hi);
#pragma unroll
        for (int kk = 0; kk < KCH; kk += 32) {
            const int p = (kk >> 5) & 1;
            if (kk + 32 < KCH) {
                load_a(abuf[p ^ 1], arowc, kk + 32, hi);
#pragma unroll
                for (int t = 0; t < 4; ++t)
                    load_b(bb[p ^ 1][t], bufc + (t * 16 + ln) * KCH, kk + 32, hi);
            }
#pragma unroll
            for (int t = 0; t < 4; ++t)
                acc[t] = wmma_bf16(abuf[p], bb[p][t], acc[t]);
        }
        if (c + 1 < NCHUNK) { async_wait0(); __syncthreads(); }
    }
    if (valid) epilogue(acc, mbase, nbase, ln, hi);
}

// ---------------- QKV GEMM: scatter q/k/v into per-head layout -------------
__global__ __launch_bounds__(256) void qkv_gemm(const unsigned short* __restrict__ xb,
                                                const unsigned short* __restrict__ wb,
                                                unsigned short* __restrict__ qb,
                                                unsigned short* __restrict__ kb,
                                                unsigned short* __restrict__ vb) {
    extern __shared__ char smem[];
    gemm_body(xb, wb, smem,
        [&](v8f (&acc)[4], int mbase, int nbase, int ln, int hi) {
#pragma unroll
            for (int t = 0; t < 4; ++t) {
                const int o = nbase + t * 16 + ln;
                const int which = o >> 10, rem = o & 1023;
                const int h = rem >> 6, d = rem & 63;
                unsigned short* dst = (which == 0) ? qb : (which == 1) ? kb : vb;
#pragma unroll
                for (int r = 0; r < 8; ++r) {
                    const int m = mbase + r + 8 * hi;
                    if (m < MROWS) {
                        const int bi = m / NSEQ, tok = m - bi * NSEQ;
                        dst[((size_t)(bi * HEADS + h) * NSEQ + tok) * DH + d] =
                            f2bf(acc[t][r]);
                    }
                }
            }
        });
}

// ---------------- output projection GEMM + bias, fp32 out ------------------
__global__ __launch_bounds__(256) void out_gemm(const unsigned short* __restrict__ ab,
                                                const unsigned short* __restrict__ wb,
                                                const float* __restrict__ bias,
                                                float* __restrict__ out) {
    extern __shared__ char smem[];
    gemm_body(ab, wb, smem,
        [&](v8f (&acc)[4], int mbase, int nbase, int ln, int hi) {
#pragma unroll
            for (int t = 0; t < 4; ++t) {
                const int o = nbase + t * 16 + ln;
                const float bo = bias[o];
#pragma unroll
                for (int r = 0; r < 8; ++r) {
                    const int m = mbase + r + 8 * hi;
                    if (m < MROWS) out[(size_t)m * DIM + o] = acc[t][r] + bo;
                }
            }
        });
}

// ---------------- divided spatial attention (one block per b*H*f group) ----
__global__ __launch_bounds__(256) void spatial_attn(const unsigned short* __restrict__ qb,
                                                    const unsigned short* __restrict__ kb,
                                                    const unsigned short* __restrict__ vb,
                                                    unsigned short* __restrict__ attnb) {
    extern __shared__ char smem[];
    unsigned short* Qs = (unsigned short*)smem;           // QP x 64
    unsigned short* Ks = Qs + QP * DH;                    // KP x 64
    unsigned short* Vt = Ks + KP * DH;                    // 64 x KP (transposed)
    unsigned short* Ps = Vt + DH * KP;                    // 8 waves x 16 x KP

    const int g  = blockIdx.x;            // 0..BH*FRM-1
    const int bh = g >> 4, fi = g & 15;
    const int bi = bh >> 4, h = bh & 15;
    const unsigned short* Qg = qb + (size_t)bh * NSEQ * DH;
    const unsigned short* Kg = kb + (size_t)bh * NSEQ * DH;
    const unsigned short* Vg = vb + (size_t)bh * NSEQ * DH;
    const int tid = threadIdx.x;

    // K rows 0..196 async (row 0 = cls token); pad rows 197..223 zeroed
    for (int c = tid; c < (NTOK + 1) * 8; c += 256) {
        const int j = c >> 3, ch = c & 7;
        const unsigned short* src = (j == 0)
            ? Kg + ch * 8
            : Kg + (size_t)(1 + fi * NTOK + (j - 1)) * DH + ch * 8;
        async_cp16(src, Ks + j * DH + ch * 8);
    }
    for (int c = tid; c < (KP - NTOK - 1) * 8; c += 256) {
        const int j = NTOK + 1 + (c >> 3), ch = c & 7;
        *reinterpret_cast<u32x4*>(Ks + j * DH + ch * 8) = u32x4{0,0,0,0};
    }
    // Q rows 0..195 async; pad rows 196..207 zeroed
    for (int c = tid; c < NTOK * 8; c += 256) {
        const int m = c >> 3, ch = c & 7;
        async_cp16(Qg + (size_t)(1 + fi * NTOK + m) * DH + ch * 8,
                   Qs + m * DH + ch * 8);
    }
    for (int c = tid; c < (QP - NTOK) * 8; c += 256) {
        const int m = NTOK + (c >> 3), ch = c & 7;
        *reinterpret_cast<u32x4*>(Qs + m * DH + ch * 8) = u32x4{0,0,0,0};
    }
    // V transposed: Vt[d][j] = V[j][d]   (element-wise; scatter, no async)
    for (int c = tid; c < DH * KP; c += 256) {
        const int d = c / KP, j = c - d * KP;
        unsigned short val = 0;
        if (j == 0)         val = Vg[d];
        else if (j <= NTOK) val = Vg[(size_t)(1 + fi * NTOK + (j - 1)) * DH + d];
        Vt[d * KP + j] = val;
    }
    async_wait0();
    __syncthreads();

    const int lane = tid & 31, wave = tid >> 5;
    const int ln = lane & 15, hi = lane >> 4;
    const float scale = 0.125f;               // 1/sqrt(64)

    for (int mt = wave; mt < QP / 16; mt += 8) {
        // ---- S = Q K^T (16 x KP), f32 accum ----
        v8f acc[KP / 16];
#pragma unroll
        for (int t = 0; t < KP / 16; ++t) acc[t] = v8f{0,0,0,0,0,0,0,0};
        const unsigned short* qrow = Qs + (mt * 16 + ln) * DH;
#pragma unroll
        for (int kk = 0; kk < DH; kk += 32) {
            BFrag16 a; load_a(a, qrow, kk, hi);
#pragma unroll
            for (int t = 0; t < KP / 16; ++t) {
                BFrag16 b; load_b(b, Ks + (t * 16 + ln) * DH, kk, hi);
                acc[t] = wmma_bf16(a, b, acc[t]);
            }
        }
        // ---- row softmax (rows live in 16-lane halves per C layout) ----
        unsigned short* pbase = Ps + (size_t)wave * 16 * KP;
#pragma unroll
        for (int r = 0; r < 8; ++r) {
            float vals[KP / 16];
            float mx = -3.4e38f;
#pragma unroll
            for (int t = 0; t < KP / 16; ++t) {
                const int col = t * 16 + ln;
                float s = (col < NTOK + 1) ? acc[t][r] * scale : -3.4e38f;
                vals[t] = s; mx = fmaxf(mx, s);
            }
#pragma unroll
            for (int o = 8; o >= 1; o >>= 1) mx = fmaxf(mx, __shfl_xor(mx, o, 32));
            float sum = 0.f;
#pragma unroll
            for (int t = 0; t < KP / 16; ++t) {
                float e = __expf(vals[t] - mx); vals[t] = e; sum += e;
            }
#pragma unroll
            for (int o = 8; o >= 1; o >>= 1) sum += __shfl_xor(sum, o, 32);
            const float inv = 1.0f / sum;
            unsigned short* prow = pbase + (r + 8 * hi) * KP;
#pragma unroll
            for (int t = 0; t < KP / 16; ++t) prow[t * 16 + ln] = f2bf(vals[t] * inv);
        }
        asm volatile("s_wait_dscnt 0" ::: "memory");   // P stores visible to own-wave loads
        // ---- O = P V (16 x 64) ----
#pragma unroll
        for (int dt = 0; dt < 4; ++dt) {
            v8f oc = v8f{0,0,0,0,0,0,0,0};
            const unsigned short* prow = pbase + ln * KP;
#pragma unroll
            for (int ks = 0; ks < KP; ks += 32) {
                BFrag16 a; load_a(a, prow, ks, hi);
                BFrag16 b; load_b(b, Vt + (dt * 16 + ln) * KP, ks, hi);
                oc = wmma_bf16(a, b, oc);
            }
#pragma unroll
            for (int r = 0; r < 8; ++r) {
                const int mg = mt * 16 + r + 8 * hi;
                if (mg < NTOK) {
                    const int tok = 1 + fi * NTOK + mg;
                    attnb[((size_t)bi * NSEQ + tok) * DIM + h * DH + dt * 16 + ln] =
                        f2bf(oc[r]);
                }
            }
        }
    }
}

// ---------------- cls-token attention (tiny; VALU two-pass softmax) --------
__global__ __launch_bounds__(256) void cls_attn(const unsigned short* __restrict__ qb,
                                                const unsigned short* __restrict__ kb,
                                                const unsigned short* __restrict__ vb,
                                                unsigned short* __restrict__ attnb) {
    __shared__ float qsh[DH];
    __shared__ float ssh[NSEQ];
    __shared__ float red[256];
    __shared__ float osh[DH];
    const int bh = blockIdx.x, bi = bh >> 4, h = bh & 15;
    const int tid = threadIdx.x;
    const unsigned short* Kg = kb + (size_t)bh * NSEQ * DH;
    const unsigned short* Vg = vb + (size_t)bh * NSEQ * DH;
    if (tid < DH) { qsh[tid] = bf2f(qb[(size_t)bh * NSEQ * DH + tid]); osh[tid] = 0.f; }
    __syncthreads();

    float lmax = -3.4e38f;
    for (int j = tid; j < NSEQ; j += 256) {
        const unsigned short* kr = Kg + (size_t)j * DH;
        float s = 0.f;
        for (int d = 0; d < DH; ++d) s += qsh[d] * bf2f(kr[d]);
        s *= 0.125f; ssh[j] = s; lmax = fmaxf(lmax, s);
    }
    red[tid] = lmax; __syncthreads();
    for (int s = 128; s > 0; s >>= 1) {
        if (tid < s) red[tid] = fmaxf(red[tid], red[tid + s]);
        __syncthreads();
    }
    const float mx = red[0]; __syncthreads();
    float lsum = 0.f;
    for (int j = tid; j < NSEQ; j += 256) {
        const float e = __expf(ssh[j] - mx); ssh[j] = e; lsum += e;
    }
    red[tid] = lsum; __syncthreads();
    for (int s = 128; s > 0; s >>= 1) {
        if (tid < s) red[tid] += red[tid + s];
        __syncthreads();
    }
    const float inv = 1.0f / red[0];
    float o[DH];
    for (int d = 0; d < DH; ++d) o[d] = 0.f;
    for (int j = tid; j < NSEQ; j += 256) {
        const float p = ssh[j] * inv;
        const unsigned short* vr = Vg + (size_t)j * DH;
        for (int d = 0; d < DH; ++d) o[d] += p * bf2f(vr[d]);
    }
    for (int d = 0; d < DH; ++d) atomicAdd(&osh[d], o[d]);
    __syncthreads();
    if (tid < DH)
        attnb[(size_t)bi * NSEQ * DIM + h * DH + tid] = f2bf(osh[tid]);
}

// ---------------- launcher --------------------------------------------------
extern "C" void kernel_launch(void* const* d_in, const int* in_sizes, int n_in,
                              void* d_out, int out_size, void* d_ws, size_t ws_size,
                              hipStream_t stream) {
    const float* x     = (const float*)d_in[0];
    const float* wqkv  = (const float*)d_in[1];
    const float* wout  = (const float*)d_in[2];
    const float* bout  = (const float*)d_in[3];
    float*       out   = (float*)d_out;

    // workspace carve (bytes, 256-aligned)
    char* ws = (char*)d_ws;
    size_t off = 0;
    auto carve = [&](size_t bytes) { char* p = ws + off; off += (bytes + 255) & ~size_t(255); return p; };
    unsigned short* xb    = (unsigned short*)carve((size_t)MROWS * DIM * 2);
    unsigned short* wqkvb = (unsigned short*)carve((size_t)QKVN * DIM * 2);
    unsigned short* woutb = (unsigned short*)carve((size_t)DIM * DIM * 2);
    unsigned short* qb    = (unsigned short*)carve((size_t)BH * NSEQ * DH * 2);
    unsigned short* kb    = (unsigned short*)carve((size_t)BH * NSEQ * DH * 2);
    unsigned short* vb    = (unsigned short*)carve((size_t)BH * NSEQ * DH * 2);
    unsigned short* attnb = (unsigned short*)carve((size_t)MROWS * DIM * 2);

    // 1) fp32 -> bf16
    cvt_bf16<<<4096, 256, 0, stream>>>(x,    xb,    MROWS * DIM);
    cvt_bf16<<<2048, 256, 0, stream>>>(wqkv, wqkvb, QKVN * DIM);
    cvt_bf16<<<1024, 256, 0, stream>>>(wout, woutb, DIM * DIM);

    const size_t gemm_lds = (size_t)2 * 64 * KCH * 2;   // 64KB double-buffered B

    // 2) qkv projection (WMMA, async-LDS staged weights)
    qkv_gemm<<<dim3((MT + 7) / 8, QKVN / 64), 256, gemm_lds, stream>>>(xb, wqkvb, qb, kb, vb);

    // 3) divided attention
    const size_t attn_lds = (size_t)(QP * DH + KP * DH + DH * KP + 8 * 16 * KP) * 2;
    spatial_attn<<<BH * FRM, 256, attn_lds, stream>>>(qb, kb, vb, attnb);
    cls_attn<<<BH, 256, 0, stream>>>(qb, kb, vb, attnb);

    // 4) output projection + bias (WMMA, fp32 out)
    out_gemm<<<dim3((MT + 7) / 8, DIM / 64), 256, gemm_lds, stream>>>(attnb, woutb, bout, out);
}
